// MoE_58042188038168
// MI455X (gfx1250) — compile-verified, matrix operand order checked by reading, not run on hardware
//
#include <hip/hip_runtime.h>
#include <hip/hip_bf16.h>

// ---------------- problem constants ----------------
#define T_TOK   2048      // BATCH*SEQ
#define DIM     1024
#define NEXP    8
#define TOPK    2
#define INTER   1408
#define SINTER  2816      // shared-expert inter dim
// padded LDS strides (halves). 1040h = 2080B = 520 dwords ≡ 8 (mod 64 banks),
// 144h = 288B = 72 dwords ≡ 8 (mod 64): lane l16 -> bank group 8*l16+4*hi,
// all 32 lanes of a ds_load_b128 hit distinct banks.
#define XSS     1040
#define HSS     144

typedef __attribute__((ext_vector_type(16))) _Float16 v16h;
typedef __attribute__((ext_vector_type(8)))  _Float16 v8h;
typedef __attribute__((ext_vector_type(8)))  float    v8f;

// ---------------- WMMA / async helpers ----------------
__device__ __forceinline__ v8f wmma_f16(v16h a, v16h b, v8f c) {
  // D = A(16x32,f16) x B(32x16,f16) + C(16x16,f32)
  return __builtin_amdgcn_wmma_f32_16x16x32_f16(
      /*neg_a=*/false, a, /*neg_b=*/false, b,
      /*c_mod=*/(short)0, c, /*reuse_a=*/false, /*reuse_b=*/false);
}

// CDNA5 async global->LDS copy, 16B per lane (ISA §10, GLOBAL_LOAD_ASYNC_TO_LDS_B128).
// the LDS destination address comes from the VDST VGPR; flat-pointer low 32 bits are the LDS offset.
__device__ __forceinline__ void async_ld_b128(unsigned lds_off, const void* g) {
  asm volatile("global_load_async_to_lds_b128 %0, %1, off"
               :: "v"(lds_off), "v"((unsigned long long)(uintptr_t)g)
               : "memory");
}
__device__ __forceinline__ void wait_async0() {
  asm volatile("s_wait_asynccnt 0x0" ::: "memory");
}

// Load a 16x32 (f16) fragment from a row-major buffer.
//  A: lane&15 selects row M; per-lane K = {hi*8+0..7} ∪ {16+hi*8+0..7} (ISA 7.12.2).
//  B (32x16, w stored [out,in] row-major): lane&15 selects column N, contiguous in K.
__device__ __forceinline__ v16h load_frag(const _Float16* __restrict__ base,
                                          int ld, int l16, int hi) {
  const _Float16* p = base + (size_t)l16 * ld + hi * 8;
  v8h lo = *(const v8h*)(p);
  v8h up = *(const v8h*)(p + 16);
  v16h f;
#pragma unroll
  for (int j = 0; j < 8; ++j) { f[j] = lo[j]; f[j + 8] = up[j]; }
  return f;
}

__device__ __forceinline__ float silu(float u) {
  return u / (1.0f + __expf(-u));
}

// ---------------- phase 0: zero expert counters ----------------
__global__ void moe_zero_counts(int* __restrict__ counts) {
  if (threadIdx.x < NEXP) counts[threadIdx.x] = 0;
}

// ---------------- phase 1: f32 -> f16 conversion ----------------
__global__ void moe_f32_to_f16(const float* __restrict__ in,
                               _Float16* __restrict__ out, size_t n) {
  size_t i = (size_t)blockIdx.x * blockDim.x + threadIdx.x;
  size_t stride = (size_t)gridDim.x * blockDim.x;
  for (; i < n; i += stride) out[i] = (_Float16)in[i];
}

// ---------------- phase 2: gating ----------------
__global__ __launch_bounds__(256) void moe_gate(
    const float* __restrict__ x, const float* __restrict__ gw,
    const float* __restrict__ gb, float* __restrict__ probs_out,
    int* __restrict__ idx_out, int* __restrict__ counts,
    int* __restrict__ lists, float* __restrict__ scale) {
  const int wv = threadIdx.x >> 5;
  const int lane = threadIdx.x & 31;
  const int t = blockIdx.x * 8 + wv;
  if (t >= T_TOK) return;

  const float* xr = x + (size_t)t * DIM;
  float acc[NEXP];
#pragma unroll
  for (int e = 0; e < NEXP; ++e) acc[e] = 0.0f;
  for (int k = lane; k < DIM; k += 32) {
    float xv = xr[k];
#pragma unroll
    for (int e = 0; e < NEXP; ++e) acc[e] += xv * gw[e * DIM + k];
  }
#pragma unroll
  for (int e = 0; e < NEXP; ++e) {
    float v = acc[e];
#pragma unroll
    for (int s = 16; s > 0; s >>= 1) v += __shfl_xor(v, s, 32);
    acc[e] = v;
  }
  if (lane == 0) {
    float sc[NEXP], m = -1e30f;
#pragma unroll
    for (int e = 0; e < NEXP; ++e) { sc[e] = acc[e] + gb[e]; m = fmaxf(m, sc[e]); }
    float sum = 0.0f;
#pragma unroll
    for (int e = 0; e < NEXP; ++e) { sc[e] = __expf(sc[e] - m); sum += sc[e]; }
    const float inv = 1.0f / sum;
    float psum = 0.0f;
#pragma unroll
    for (int e = 0; e < NEXP; ++e) { sc[e] *= inv; psum += sc[e]; }
    int i0 = 0;
#pragma unroll
    for (int e = 1; e < NEXP; ++e) if (sc[e] > sc[i0]) i0 = e;
    int i1 = (i0 == 0) ? 1 : 0;
#pragma unroll
    for (int e = 0; e < NEXP; ++e) if (e != i0 && sc[e] > sc[i1]) i1 = e;
#pragma unroll
    for (int e = 0; e < NEXP; ++e) probs_out[(size_t)t * NEXP + e] = sc[e];
    idx_out[t * 2 + 0] = i0;
    idx_out[t * 2 + 1] = i1;
    scale[t] = psum;  // == probs.sum(-1), faithful to reference
    int p0 = atomicAdd(&counts[i0], 1); lists[i0 * T_TOK + p0] = t;
    int p1 = atomicAdd(&counts[i1], 1); lists[i1 * T_TOK + p1] = t;
  }
}

// ---------------- phase 3: shared-expert fused SwiGLU (M=16 tiles) ----------------
__global__ __launch_bounds__(256) void moe_shared_ffn(
    const _Float16* __restrict__ xh, const _Float16* __restrict__ sw1h,
    const _Float16* __restrict__ sw3h, const _Float16* __restrict__ sw2h,
    const float* __restrict__ sb1, const float* __restrict__ sb3,
    const float* __restrict__ sb2, float* __restrict__ y) {
  __shared__ __align__(16) _Float16 xs[16 * XSS];
  __shared__ __align__(16) _Float16 hs[16 * HSS];

  const int tile = blockIdx.x;
  const int tid = threadIdx.x;
  const int lane = tid & 31, l16 = lane & 15, hi = lane >> 4, wv = tid >> 5;

  {  // async-stage 16 x-rows: 256 threads x 64 halves (8 x 16B)
    const int row = tid >> 4;
    const int col0 = (tid & 15) * 64;
    const _Float16* g = xh + (size_t)(tile * 16 + row) * DIM + col0;
    const unsigned l = (unsigned)(uintptr_t)(xs + row * XSS + col0);
#pragma unroll
    for (int i = 0; i < 8; ++i) async_ld_b128(l + i * 16, g + i * 8);
    wait_async0();
  }
  __syncthreads();

  v8f oacc[8];
#pragma unroll
  for (int t = 0; t < 8; ++t) { v8f z = {}; oacc[t] = z; }

  for (int c = 0; c < SINTER / 128; ++c) {  // 22 chunks of 128 inter-cols
    const int nG = c * 128 + wv * 16;
    v8f a1 = {}, a3 = {};
    const _Float16* bw1 = sw1h + (size_t)nG * DIM;
    const _Float16* bw3 = sw3h + (size_t)nG * DIM;
    for (int kk = 0; kk < DIM; kk += 32) {
      v16h af = load_frag(xs + kk, XSS, l16, hi);
      v16h f1 = load_frag(bw1 + kk, DIM, l16, hi);
      v16h f3 = load_frag(bw3 + kk, DIM, l16, hi);
      a1 = wmma_f16(af, f1, a1);
      a3 = wmma_f16(af, f3, a3);
    }
    {
      const float bb1 = sb1[nG + l16];
      const float bb3 = sb3[nG + l16];
#pragma unroll
      for (int r = 0; r < 8; ++r) {
        float g = silu(a1[r] + bb1) * (a3[r] + bb3);
        hs[(r + hi * 8) * HSS + wv * 16 + l16] = (_Float16)g;
      }
    }
    __syncthreads();
#pragma unroll
    for (int t = 0; t < 8; ++t) {
      const int n = (wv * 8 + t) * 16;
      const _Float16* bw2 = sw2h + (size_t)n * SINTER + c * 128;
#pragma unroll
      for (int kk = 0; kk < 128; kk += 32) {
        v16h af = load_frag(hs + kk, HSS, l16, hi);
        v16h bf = load_frag(bw2 + kk, SINTER, l16, hi);
        oacc[t] = wmma_f16(af, bf, oacc[t]);
      }
    }
    __syncthreads();
  }

#pragma unroll
  for (int t = 0; t < 8; ++t) {
    const int col = (wv * 8 + t) * 16 + l16;
    const float bb = sb2[col];
#pragma unroll
    for (int r = 0; r < 8; ++r) {
      const int tok = tile * 16 + (r + hi * 8);
      y[(size_t)tok * DIM + col] = oacc[t][r] + bb;  // first writer of y
    }
  }
}

// ---------------- phase 4: routed experts, fused SwiGLU (M=32 tiles) ----------------
// Each B fragment feeds two WMMAs (two 16-row A tiles) -> 32 FLOP per L2 byte.
__global__ __launch_bounds__(256) void moe_routed_ffn(
    const _Float16* __restrict__ xh, const _Float16* __restrict__ w1h,
    const _Float16* __restrict__ w3h, const _Float16* __restrict__ w2h,
    const float* __restrict__ b1, const float* __restrict__ b3,
    const float* __restrict__ b2, const int* __restrict__ counts,
    const int* __restrict__ lists, const float* __restrict__ scale,
    float* __restrict__ y) {
  extern __shared__ __align__(16) char smem[];           // ~74 KB dynamic LDS
  _Float16* xs = (_Float16*)smem;                        // 32 * XSS halves
  _Float16* hs = (_Float16*)(smem + 32 * XSS * 2);       // 32 * HSS halves
  int* toks    = (int*)(smem + 32 * XSS * 2 + 32 * HSS * 2);

  const int e = blockIdx.x >> 6;      // 64 tiles of 32 tokens per expert
  const int tile = blockIdx.x & 63;
  const int cnt = counts[e];
  if (tile * 32 >= cnt) return;

  const int tid = threadIdx.x;
  const int lane = tid & 31, l16 = lane & 15, hi = lane >> 4, wv = tid >> 5;

  if (tid < 32) {
    const int p = tile * 32 + tid;
    toks[tid] = (p < cnt) ? lists[e * T_TOK + p] : -1;
  }
  __syncthreads();

  {  // async gather-stage 32 token rows: 256 threads x 128 halves (16 x 16B)
    const int row = tid >> 3;
    const int col0 = (tid & 7) * 128;
    const int tsrc = (toks[row] >= 0) ? toks[row] : toks[0];
    const _Float16* g = xh + (size_t)tsrc * DIM + col0;
    const unsigned l = (unsigned)(uintptr_t)(xs + row * XSS + col0);
#pragma unroll
    for (int i = 0; i < 16; ++i) async_ld_b128(l + i * 16, g + i * 8);
    wait_async0();
  }
  __syncthreads();

  v8f oacc[16];  // 8 n-tiles x 2 m-tiles per wave
#pragma unroll
  for (int t = 0; t < 16; ++t) { v8f z = {}; oacc[t] = z; }

  for (int c = 0; c < INTER / 128; ++c) {  // 11 chunks of 128 inter-cols
    const int nG = c * 128 + wv * 16;
    v8f a1m0 = {}, a3m0 = {}, a1m1 = {}, a3m1 = {};
    const _Float16* bw1 = w1h + ((size_t)e * INTER + nG) * DIM;
    const _Float16* bw3 = w3h + ((size_t)e * INTER + nG) * DIM;
    for (int kk = 0; kk < DIM; kk += 32) {
      v16h f1 = load_frag(bw1 + kk, DIM, l16, hi);
      v16h f3 = load_frag(bw3 + kk, DIM, l16, hi);
      v16h am0 = load_frag(xs + kk, XSS, l16, hi);
      v16h am1 = load_frag(xs + 16 * XSS + kk, XSS, l16, hi);
      a1m0 = wmma_f16(am0, f1, a1m0);
      a3m0 = wmma_f16(am0, f3, a3m0);
      a1m1 = wmma_f16(am1, f1, a1m1);
      a3m1 = wmma_f16(am1, f3, a3m1);
    }
    {
      const float bb1 = b1[(size_t)e * INTER + nG + l16];
      const float bb3 = b3[(size_t)e * INTER + nG + l16];
#pragma unroll
      for (int r = 0; r < 8; ++r) {
        float g0 = silu(a1m0[r] + bb1) * (a3m0[r] + bb3);
        float g1 = silu(a1m1[r] + bb1) * (a3m1[r] + bb3);
        hs[(r + hi * 8) * HSS + wv * 16 + l16] = (_Float16)g0;
        hs[(16 + r + hi * 8) * HSS + wv * 16 + l16] = (_Float16)g1;
      }
    }
    __syncthreads();
#pragma unroll
    for (int t = 0; t < 8; ++t) {
      const int n = (wv * 8 + t) * 16;
      const _Float16* bw2 = w2h + ((size_t)e * DIM + n) * INTER + c * 128;
#pragma unroll
      for (int kk = 0; kk < 128; kk += 32) {
        v16h bf = load_frag(bw2 + kk, INTER, l16, hi);
        v16h h0 = load_frag(hs + kk, HSS, l16, hi);
        v16h h1 = load_frag(hs + 16 * HSS + kk, HSS, l16, hi);
        oacc[t * 2 + 0] = wmma_f16(h0, bf, oacc[t * 2 + 0]);
        oacc[t * 2 + 1] = wmma_f16(h1, bf, oacc[t * 2 + 1]);
      }
    }
    __syncthreads();
  }

#pragma unroll
  for (int t = 0; t < 8; ++t) {
    const int col = (wv * 8 + t) * 16 + l16;
    const float bb = b2[(size_t)e * DIM + col];
#pragma unroll
    for (int m = 0; m < 2; ++m) {
#pragma unroll
      for (int r = 0; r < 8; ++r) {
        const int row = m * 16 + r + hi * 8;
        if (tile * 32 + row < cnt) {
          const int tok = toks[row];
          atomicAdd(&y[(size_t)tok * DIM + col],
                    (oacc[t * 2 + m][r] + bb) * scale[tok]);
        }
      }
    }
  }
}

// ---------------- host launcher ----------------
extern "C" void kernel_launch(void* const* d_in, const int* in_sizes, int n_in,
                              void* d_out, int out_size, void* d_ws, size_t ws_size,
                              hipStream_t stream) {
  (void)in_sizes; (void)n_in; (void)out_size; (void)ws_size;

  const float* x   = (const float*)d_in[0];
  const float* gw  = (const float*)d_in[1];
  const float* gb  = (const float*)d_in[2];
  const float* w1  = (const float*)d_in[3];
  const float* b1  = (const float*)d_in[4];
  const float* w2  = (const float*)d_in[5];
  const float* b2  = (const float*)d_in[6];
  const float* w3  = (const float*)d_in[7];
  const float* b3  = (const float*)d_in[8];
  const float* sw1 = (const float*)d_in[9];
  const float* sb1 = (const float*)d_in[10];
  const float* sw2 = (const float*)d_in[11];
  const float* sb2 = (const float*)d_in[12];
  const float* sw3 = (const float*)d_in[13];
  const float* sb3 = (const float*)d_in[14];

  // output layout: probs [T,8] | idx [T,2] (int bits) | y [T,D]
  float* out_f = (float*)d_out;
  float* probs = out_f;
  int*   idx   = (int*)(out_f + (size_t)T_TOK * NEXP);
  float* y     = out_f + (size_t)T_TOK * NEXP + (size_t)T_TOK * TOPK;

  // workspace carve-out (~91 MB; f16 weights stay L2-resident at 192 MB)
  char* ws = (char*)d_ws;
  size_t off = 0;
  auto take = [&](size_t bytes) -> char* {
    char* p = ws + off;
    off = (off + bytes + 255) & ~(size_t)255;
    return p;
  };
  int*      counts = (int*)take(NEXP * sizeof(int));
  int*      lists  = (int*)take((size_t)NEXP * T_TOK * sizeof(int));
  float*    scale  = (float*)take((size_t)T_TOK * sizeof(float));
  _Float16* xh     = (_Float16*)take((size_t)T_TOK * DIM * 2);
  _Float16* w1h    = (_Float16*)take((size_t)NEXP * INTER * DIM * 2);
  _Float16* w3h    = (_Float16*)take((size_t)NEXP * INTER * DIM * 2);
  _Float16* w2h    = (_Float16*)take((size_t)NEXP * DIM * INTER * 2);
  _Float16* sw1h   = (_Float16*)take((size_t)SINTER * DIM * 2);
  _Float16* sw3h   = (_Float16*)take((size_t)SINTER * DIM * 2);
  _Float16* sw2h   = (_Float16*)take((size_t)DIM * SINTER * 2);

  auto cgrid = [](size_t n) -> unsigned {
    return (unsigned)((n + 256 * 8 - 1) / (256 * 8));  // 8 elems/thread
  };

  moe_zero_counts<<<1, 32, 0, stream>>>(counts);

  moe_f32_to_f16<<<cgrid((size_t)T_TOK * DIM), 256, 0, stream>>>(x, xh, (size_t)T_TOK * DIM);
  moe_f32_to_f16<<<cgrid((size_t)NEXP * INTER * DIM), 256, 0, stream>>>(w1, w1h, (size_t)NEXP * INTER * DIM);
  moe_f32_to_f16<<<cgrid((size_t)NEXP * INTER * DIM), 256, 0, stream>>>(w3, w3h, (size_t)NEXP * INTER * DIM);
  moe_f32_to_f16<<<cgrid((size_t)NEXP * DIM * INTER), 256, 0, stream>>>(w2, w2h, (size_t)NEXP * DIM * INTER);
  moe_f32_to_f16<<<cgrid((size_t)SINTER * DIM), 256, 0, stream>>>(sw1, sw1h, (size_t)SINTER * DIM);
  moe_f32_to_f16<<<cgrid((size_t)SINTER * DIM), 256, 0, stream>>>(sw3, sw3h, (size_t)SINTER * DIM);
  moe_f32_to_f16<<<cgrid((size_t)DIM * SINTER), 256, 0, stream>>>(sw2, sw2h, (size_t)DIM * SINTER);

  moe_gate<<<T_TOK / 8, 256, 0, stream>>>(x, gw, gb, probs, idx, counts, lists, scale);

  // shared MLP writes y (plain stores) first; routed experts then atomically add
  moe_shared_ffn<<<T_TOK / 16, 256, 0, stream>>>(xh, sw1h, sw3h, sw2h, sb1, sb3, sb2, y);

  const size_t smem_routed = (size_t)32 * XSS * 2 + (size_t)32 * HSS * 2 + 32 * sizeof(int);
  moe_routed_ffn<<<NEXP * (T_TOK / 32), 256, smem_routed, stream>>>(
      xh, w1h, w3h, w2h, b1, b3, b2, counts, lists, scale, y);
}
